// SocialLSTM_19670950216139
// MI455X (gfx1250) — matrix operand order repaired and checked by reading, not compiled, not run on hardware
//
#include <hip/hip_runtime.h>
#include <hip/hip_bf16.h>

// ---------------------------------------------------------------------------
// Social-LSTM on MI455X (gfx1250, wave32, WMMA), fused version.
//  - enc_all_kernel: all 8 encoder LSTM steps in one launch; h lives in LDS,
//    c lives in registers; 512 independent blocks of 16 pedestrians.
//  - dec_all_kernel: all 12 decoder steps in one launch; one block per scene
//    (128 peds = 1024 threads = 32 waves). proj GEMM -> social gather ->
//    gate GEMM -> LSTM -> output projection, separated by block barriers.
//  All GEMMs use v_wmma_f32_16x16x32_f16 (f16 in, f32 accumulate).
//  Scene-state staging uses global_load_async_to_lds_b128 (ASYNCcnt).
// ---------------------------------------------------------------------------

#define OBS_LEN 8
#define PRED_LEN 12
#define EMB 64
#define HID 128
#define POOL 64
#define SCENES 64
#define PEDS 128
#define TOTAL (SCENES * PEDS)
#define NSF 2.0f

typedef __attribute__((ext_vector_type(16))) _Float16 v16h;
typedef __attribute__((ext_vector_type(8)))  _Float16 v8h;
typedef __attribute__((ext_vector_type(8)))  float    v8f;

__device__ __forceinline__ float sigmoidf(float x) {
    return 1.0f / (1.0f + __expf(-x));
}

__device__ __forceinline__ v8f wmma_f16(v16h a, v16h b, v8f c) {
    return __builtin_amdgcn_wmma_f32_16x16x32_f16(
        false, a, false, b, (short)0, c, false, false);
}

// Async 16-byte global->LDS copy (per-lane), tracked by ASYNCcnt.
// VDST = LDS byte offset (generic shared addr low 32 bits, ISA 10.2),
// VADDR = 64-bit global address (GV mode).
__device__ __forceinline__ void async_g2l_b128(_Float16* lds_dst,
                                               const _Float16* src) {
    unsigned int loff = (unsigned int)(unsigned long long)lds_dst;
    asm volatile("global_load_async_to_lds_b128 %0, %1, off"
                 : : "v"(loff), "v"(src) : "memory");
}

__device__ __forceinline__ void wait_asynccnt0() {
    asm volatile("s_wait_asynccnt 0x0" ::: "memory");
}

// A fragment (16x32 f16, MxK) from row-major f16, leading dim ld.
// Per ISA 7.12.2 the per-lane halves are two groups of 8 contiguous K values
// => two 16B loads per lane (LDS: ds_load_b128, global: global_load_b128).
__device__ __forceinline__ v16h load_a_frag(const _Float16* X, int ld,
                                            int ktile, int lane) {
    int m  = lane & 15;
    int kb = ktile * 32 + ((lane & 16) ? 8 : 0);
    const _Float16* p = X + m * ld + kb;
    v8h lo = *(const v8h*)(p);
    v8h hi = *(const v8h*)(p + 16);
    return __builtin_shufflevector(lo, hi, 0, 1, 2, 3, 4, 5, 6, 7,
                                           8, 9, 10, 11, 12, 13, 14, 15);
}

// B fragment from pre-packed buffer: one contiguous 32B v16h per lane.
__device__ __forceinline__ v16h load_b_frag(const _Float16* P, int Kt,
                                            int nt, int kt, int lane) {
    return *((const v16h*)P + ((nt * Kt + kt) * 32 + lane));
}

// ---------------------------------------------------------------------------
// Pack f32 weights into f16 WMMA B-fragment order.
// mode 0: B(k,n) = W[n*K + k]                         (gates = x @ W^T)
// mode 1: B(k,n) = W_grid[(n%64)*2048 + (n/64)*128+k] (social projection)
// ---------------------------------------------------------------------------
__global__ __launch_bounds__(256) void pack_b_kernel(
    const float* __restrict__ W, _Float16* __restrict__ dst,
    int K, int N, int mode) {
    int idx = blockIdx.x * 256 + threadIdx.x;
    int total = (K / 32) * (N / 16) * 512;
    if (idx >= total) return;
    int i    = idx & 15;
    int lane = (idx >> 4) & 31;
    int tile = idx >> 9;
    int Kt = K / 32;
    int kt = tile % Kt;
    int nt = tile / Kt;
    int v = i >> 1, hb = i & 1;
    int k = kt * 32 + ((lane & 16) ? 8 : 0) + ((i < 8) ? i : 8 + i);
    int n = nt * 16 + (lane & 15);
    float val = (mode == 0) ? W[n * K + k]
                            : W[(n & 63) * 2048 + (n >> 6) * 128 + k];
    dst[idx] = (_Float16)val;
}

// ---------------------------------------------------------------------------
// Fused encoder: 8 LSTM steps, block = 16 rows, 256 threads = 8 waves.
// h kept in LDS (f16), c in registers; exported once at the end.
// ---------------------------------------------------------------------------
__global__ __launch_bounds__(256) void enc_all_kernel(
    const float* __restrict__ obs,              // (8, TOTAL, 2)
    const float* __restrict__ W_emb,
    const float* __restrict__ b_emb,
    const _Float16* __restrict__ Bih,           // packed, Kt=2, Nt=32
    const _Float16* __restrict__ Bhh,           // packed, Kt=4, Nt=32
    const float* __restrict__ b_ih,
    const float* __restrict__ b_hh,
    _Float16* __restrict__ h16,                 // (TOTAL,128) out
    float* __restrict__ c_buf) {                // (TOTAL,128) out
    __shared__ _Float16 hL[16 * HID];
    __shared__ _Float16 embs[16 * EMB];
    int tid = threadIdx.x;
    int m0 = blockIdx.x * 16;
    int w = tid >> 5, lane = tid & 31;
    int lm = lane & 15;
    int lh = (lane & 16) ? 8 : 0;
    int ncol = w * 16 + lm;

    for (int q = tid; q < 16 * HID; q += 256) hL[q] = (_Float16)0.0f;
    float creg[8];
#pragma unroll
    for (int r = 0; r < 8; ++r) creg[r] = 0.0f;

    for (int t = 0; t < OBS_LEN; ++t) {
        const float* pos_t = obs + (size_t)t * TOTAL * 2;
        for (int q = tid; q < 16 * EMB; q += 256) {
            int m = q >> 6, e = q & 63;
            float p0 = pos_t[(m0 + m) * 2 + 0];
            float p1 = pos_t[(m0 + m) * 2 + 1];
            embs[q] = (_Float16)(p0 * W_emb[2 * e] + p1 * W_emb[2 * e + 1] + b_emb[e]);
        }
        // hide next step's position-row latency behind this step's GEMMs
        if (t + 1 < OBS_LEN && tid < 16)
            __builtin_prefetch(obs + (size_t)(t + 1) * TOTAL * 2 + (m0 + tid) * 2,
                               0, 0);
        __syncthreads();   // embs ready; prev-step hL writes visible

        v16h a_e[2], a_h[4];
#pragma unroll
        for (int kt = 0; kt < 2; ++kt) a_e[kt] = load_a_frag(embs, EMB, kt, lane);
#pragma unroll
        for (int kt = 0; kt < 4; ++kt) a_h[kt] = load_a_frag(hL, HID, kt, lane);

        v8f acc[4];
#pragma unroll
        for (int s = 0; s < 4; ++s) {           // gates i,f,g,o
            int nt = s * 8 + w;
            v8f cacc = {};
#pragma unroll
            for (int kt = 0; kt < 2; ++kt)
                cacc = wmma_f16(a_e[kt], load_b_frag(Bih, 2, nt, kt, lane), cacc);
#pragma unroll
            for (int kt = 0; kt < 4; ++kt)
                cacc = wmma_f16(a_h[kt], load_b_frag(Bhh, 4, nt, kt, lane), cacc);
            float bias = b_ih[nt * 16 + lm] + b_hh[nt * 16 + lm];
#pragma unroll
            for (int r = 0; r < 8; ++r) cacc[r] += bias;
            acc[s] = cacc;
        }
        __syncthreads();   // all reads of embs/hL done before rewriting hL

#pragma unroll
        for (int r = 0; r < 8; ++r) {
            int ml = r + lh;
            float iv = sigmoidf(acc[0][r]);
            float fv = sigmoidf(acc[1][r]);
            float gv = tanhf(acc[2][r]);
            float ov = sigmoidf(acc[3][r]);
            float cn = fv * creg[r] + iv * gv;
            creg[r] = cn;
            hL[ml * HID + ncol] = (_Float16)(ov * tanhf(cn));
        }
    }
    __syncthreads();
    // export state for the decoder
    for (int q = tid; q < 16 * HID; q += 256)
        h16[(size_t)m0 * HID + q] = hL[q];
#pragma unroll
    for (int r = 0; r < 8; ++r)
        c_buf[(size_t)(m0 + r + lh) * HID + ncol] = creg[r];
}

// ---------------------------------------------------------------------------
// Fused decoder: 12 steps, one block per scene (128 peds, 1024 thr, 32 waves).
// Wave w: row-tile rt = w>>2 (16 rows), column group cg = w&3.
// Per step: proj GEMM (128x1024) -> social gather+emb -> gate GEMM (128x512)
//           -> LSTM elementwise -> output projection / position update.
// ---------------------------------------------------------------------------
__global__ __launch_bounds__(1024) void dec_all_kernel(
    const float* __restrict__ obs,               // (8, TOTAL, 2)
    const _Float16* __restrict__ h16_in,         // (TOTAL,128) from encoder
    const float* __restrict__ c_in,              // (TOTAL,128) from encoder
    const _Float16* __restrict__ Bih,            // packed, Kt=4, Nt=32
    const _Float16* __restrict__ Bhh,            // packed, Kt=4, Nt=32
    const _Float16* __restrict__ Bgrd,           // packed, Kt=4, Nt=64
    const float* __restrict__ b_ih,
    const float* __restrict__ b_hh,
    const float* __restrict__ W_emb,
    const float* __restrict__ b_emb,
    const float* __restrict__ b_grid,
    const float* __restrict__ W_out,             // (2,128)
    const float* __restrict__ b_out,             // (2)
    _Float16* __restrict__ proj,                 // (TOTAL,1024) scratch
    float* __restrict__ out) {                   // (12, TOTAL, 2)
    __shared__ _Float16 hL[PEDS * HID];          // 32 KB
    __shared__ _Float16 xL[PEDS * HID];          // 32 KB
    __shared__ float px[PEDS], py[PEDS];
    int sc = blockIdx.x;
    int tid = threadIdx.x, w = tid >> 5, lane = tid & 31;
    int rt = w >> 2, cg = w & 3;
    int lm = lane & 15;
    int lh = (lane & 16) ? 8 : 0;
    int ped = tid >> 3, q8 = tid & 7;

    // ---- stage scene h into LDS via async global->LDS (ASYNCcnt) ----
    // 128*128 halves = 2048 x 16B chunks; 2 chunks per thread.
#pragma unroll
    for (int q = 0; q < 2; ++q) {
        int c16 = tid + q * 1024;                // chunk index, 8 halves each
        async_g2l_b128(hL + c16 * 8,
                       h16_in + (size_t)sc * PEDS * HID + c16 * 8);
    }
    // overlap: c state register loads while the async copy is in flight
    float creg[2][8];
#pragma unroll
    for (int u = 0; u < 2; ++u) {
        int ncol = cg * 32 + u * 16 + lm;
#pragma unroll
        for (int r = 0; r < 8; ++r) {
            int ml = rt * 16 + r + lh;
            creg[u][r] = c_in[((size_t)sc * PEDS + ml) * HID + ncol];
        }
    }
    if (tid < PEDS) {
        px[tid] = obs[(size_t)((OBS_LEN - 1) * TOTAL + sc * PEDS + tid) * 2 + 0];
        py[tid] = obs[(size_t)((OBS_LEN - 1) * TOTAL + sc * PEDS + tid) * 2 + 1];
    }
    wait_asynccnt0();
    __syncthreads();

    for (int t = 0; t < PRED_LEN; ++t) {
        // ---- proj = h @ B_grid : wave covers 16 n-tiles of its row-tile ----
        {
            v16h a[4];
#pragma unroll
            for (int kt = 0; kt < 4; ++kt)
                a[kt] = load_a_frag(hL + rt * 16 * HID, HID, kt, lane);
            for (int s = 0; s < 16; ++s) {
                int nt = cg * 16 + s;
                v8f acc = {};
#pragma unroll
                for (int kt = 0; kt < 4; ++kt)
                    acc = wmma_f16(a[kt], load_b_frag(Bgrd, 4, nt, kt, lane), acc);
                int n = nt * 16 + lm;
#pragma unroll
                for (int r = 0; r < 8; ++r)
                    proj[((size_t)sc * PEDS + rt * 16 + r + lh) * 1024 + n] =
                        (_Float16)acc[r];
            }
        }
        __threadfence();   // proj stores land in L2 before same-block reads
        __syncthreads();

        // ---- social gather + embedding: 8 threads per ped, 8 comps each ----
        {
            float xi = px[ped], yi = py[ped];
#pragma unroll
            for (int i = 0; i < 8; ++i) {
                int e = q8 * 8 + i;
                xL[ped * HID + e] =
                    (_Float16)(xi * W_emb[2 * e] + yi * W_emb[2 * e + 1] + b_emb[e]);
            }
            float acc8[8];
#pragma unroll
            for (int i = 0; i < 8; ++i) acc8[i] = b_grid[q8 * 8 + i];
            for (int j = 0; j < PEDS; ++j) {
                if (j == ped) continue;
                float dx = px[j] - xi, dy = py[j] - yi;
                if (fabsf(dx) <= NSF && fabsf(dy) <= NSF) {
                    // (rel+NS)/(2NS)*GRID == rel + 2 for NS=2, GRID=4
                    int gx = min(max((int)floorf(dx + 2.0f), 0), 3);
                    int gy = min(max((int)floorf(dy + 2.0f), 0), 3);
                    int gidx = gy * 4 + gx;
                    v8h pv = *(const v8h*)(proj + ((size_t)sc * PEDS + j) * 1024 +
                                           gidx * 64 + q8 * 8);
#pragma unroll
                    for (int i = 0; i < 8; ++i) acc8[i] += (float)pv[i];
                }
            }
#pragma unroll
            for (int i = 0; i < 8; ++i)
                xL[ped * HID + 64 + q8 * 8 + i] = (_Float16)acc8[i];
        }
        __syncthreads();

        // ---- gate GEMM: wave owns rows rt*16..+15, cols cg*32..+31/gate ----
        v8f acc[4][2];
#pragma unroll
        for (int s = 0; s < 4; ++s)
#pragma unroll
            for (int u = 0; u < 2; ++u) acc[s][u] = (v8f){};
#pragma unroll
        for (int kt = 0; kt < 4; ++kt) {
            v16h ax = load_a_frag(xL + rt * 16 * HID, HID, kt, lane);
            v16h ah = load_a_frag(hL + rt * 16 * HID, HID, kt, lane);
#pragma unroll
            for (int s = 0; s < 4; ++s)
#pragma unroll
                for (int u = 0; u < 2; ++u) {
                    int nt = s * 8 + cg * 2 + u;
                    acc[s][u] = wmma_f16(ax, load_b_frag(Bih, 4, nt, kt, lane),
                                         acc[s][u]);
                    acc[s][u] = wmma_f16(ah, load_b_frag(Bhh, 4, nt, kt, lane),
                                         acc[s][u]);
                }
        }
        __syncthreads();   // finish reading hL/xL before rewriting hL

        // ---- LSTM elementwise ----
#pragma unroll
        for (int u = 0; u < 2; ++u) {
            int ncol = cg * 32 + u * 16 + lm;
            float bi = b_ih[0 * 128 + ncol] + b_hh[0 * 128 + ncol];
            float bf = b_ih[1 * 128 + ncol] + b_hh[1 * 128 + ncol];
            float bg = b_ih[2 * 128 + ncol] + b_hh[2 * 128 + ncol];
            float bo = b_ih[3 * 128 + ncol] + b_hh[3 * 128 + ncol];
#pragma unroll
            for (int r = 0; r < 8; ++r) {
                int ml = rt * 16 + r + lh;
                float iv = sigmoidf(acc[0][u][r] + bi);
                float fv = sigmoidf(acc[1][u][r] + bf);
                float gv = tanhf(acc[2][u][r] + bg);
                float ov = sigmoidf(acc[3][u][r] + bo);
                float cn = fv * creg[u][r] + iv * gv;
                creg[u][r] = cn;
                hL[ml * HID + ncol] = (_Float16)(ov * tanhf(cn));
            }
        }
        __syncthreads();

        // ---- output projection + position update ----
        if (tid < PEDS) {
            float d0 = b_out[0], d1 = b_out[1];
#pragma unroll 8
            for (int k = 0; k < HID; ++k) {
                float hv = (float)hL[tid * HID + k];
                d0 += hv * W_out[k];
                d1 += hv * W_out[HID + k];
            }
            float nx = px[tid] + d0;
            float ny = py[tid] + d1;
            px[tid] = nx;
            py[tid] = ny;
            float* o = out + (size_t)t * TOTAL * 2 + ((size_t)sc * PEDS + tid) * 2;
            o[0] = nx;
            o[1] = ny;
        }
        __syncthreads();
    }
}

// ---------------------------------------------------------------------------
extern "C" void kernel_launch(void* const* d_in, const int* in_sizes, int n_in,
                              void* d_out, int out_size, void* d_ws, size_t ws_size,
                              hipStream_t stream) {
    const float* obs      = (const float*)d_in[0];   // (8, TOTAL, 2)
    const float* W_emb    = (const float*)d_in[2];
    const float* b_emb    = (const float*)d_in[3];
    const float* W_ih_enc = (const float*)d_in[4];
    const float* W_hh_enc = (const float*)d_in[5];
    const float* b_ih_enc = (const float*)d_in[6];
    const float* b_hh_enc = (const float*)d_in[7];
    const float* W_grid   = (const float*)d_in[8];
    const float* b_grid   = (const float*)d_in[9];
    const float* W_ih_dec = (const float*)d_in[10];
    const float* W_hh_dec = (const float*)d_in[11];
    const float* b_ih_dec = (const float*)d_in[12];
    const float* b_hh_dec = (const float*)d_in[13];
    const float* W_out    = (const float*)d_in[14];
    const float* b_out    = (const float*)d_in[15];
    float* out = (float*)d_out;

    // workspace carve-up (~23 MB)
    char* p = (char*)d_ws;
    float*    c_buf = (float*)p;     p += (size_t)TOTAL * HID * 4;   // 4 MB
    _Float16* h16   = (_Float16*)p;  p += (size_t)TOTAL * HID * 2;   // 2 MB
    _Float16* proj  = (_Float16*)p;  p += (size_t)TOTAL * 1024 * 2;  // 16 MB
    _Float16* Bih_e = (_Float16*)p;  p += (size_t)32768 * 2;   // K=64,  N=512
    _Float16* Bhh_e = (_Float16*)p;  p += (size_t)65536 * 2;   // K=128, N=512
    _Float16* Bih_d = (_Float16*)p;  p += (size_t)65536 * 2;   // K=128, N=512
    _Float16* Bhh_d = (_Float16*)p;  p += (size_t)65536 * 2;   // K=128, N=512
    _Float16* Bgrd  = (_Float16*)p;  p += (size_t)131072 * 2;  // K=128, N=1024

    // pack weights into WMMA B-fragment order
    pack_b_kernel<<<32768 / 256, 256, 0, stream>>>(W_ih_enc, Bih_e, 64, 512, 0);
    pack_b_kernel<<<65536 / 256, 256, 0, stream>>>(W_hh_enc, Bhh_e, 128, 512, 0);
    pack_b_kernel<<<65536 / 256, 256, 0, stream>>>(W_ih_dec, Bih_d, 128, 512, 0);
    pack_b_kernel<<<65536 / 256, 256, 0, stream>>>(W_hh_dec, Bhh_d, 128, 512, 0);
    pack_b_kernel<<<131072 / 256, 256, 0, stream>>>(W_grid, Bgrd, 128, 1024, 1);

    // all 8 encoder steps in one launch
    enc_all_kernel<<<TOTAL / 16, 256, 0, stream>>>(
        obs, W_emb, b_emb, Bih_e, Bhh_e, b_ih_enc, b_hh_enc, h16, c_buf);

    // all 12 decoder steps in one launch (one block per scene)
    dec_all_kernel<<<SCENES, 1024, 0, stream>>>(
        obs, h16, c_buf, Bih_d, Bhh_d, Bgrd, b_ih_dec, b_hh_dec,
        W_emb, b_emb, b_grid, W_out, b_out, proj, out);
}
